// _NonLocalBlockND_86260123173183
// MI455X (gfx1250) — matrix operand order repaired and checked by reading, not compiled, not run on hardware
//
#include <hip/hip_runtime.h>
#include <hip/hip_bf16.h>

typedef _Float16 half_t;
typedef __attribute__((ext_vector_type(16))) _Float16 v16h;
typedef __attribute__((ext_vector_type(8)))  _Float16 v8h;
typedef __attribute__((ext_vector_type(8)))  float    v8f;
typedef __attribute__((ext_vector_type(4)))  float    v4f;

#define NTOT 4096   // H*W
#define CIN  64
#define IC   32

// low 32 bits of a generic pointer into LDS == wave-relative LDS byte address
__device__ __forceinline__ uint32_t lds_off(const void* p) {
    return (uint32_t)(uintptr_t)p;
}

// ---------------------------------------------------------------------------
// Kernel 0: cast memory bank to f16 in both layouts
// mb: (32, 256) f32.  mbC[c*256+k] = mb[c][k];  mbT[k*32+c] = mb[c][k]
// ---------------------------------------------------------------------------
__global__ void init_mb_kernel(const float* __restrict__ mb,
                               half_t* __restrict__ mbC,
                               half_t* __restrict__ mbT) {
    int i = blockIdx.x * 256 + threadIdx.x;      // i < 32*256
    int c = i >> 8, k = i & 255;
    float v = mb[i];
    mbC[i] = (half_t)v;
    mbT[k * IC + c] = (half_t)v;
}

// ---------------------------------------------------------------------------
// Kernel 1: fused 1x1 conv projections (g, theta, phi), fp32 accumulate.
// Outputs:
//   thetaT (B, N, 32) f16 row-major   (Q for phase 1)
//   phiT   (B, N, 32) f16 row-major   (K for phase 1, Q for phase 2)
//   Vc     (B, 32, N) f16 c-major     (V for phase 1)
// ---------------------------------------------------------------------------
__global__ __launch_bounds__(256) void proj_kernel(
    const float* __restrict__ x,
    const float* __restrict__ g_w,  const float* __restrict__ g_b,
    const float* __restrict__ t_w,  const float* __restrict__ t_b,
    const float* __restrict__ p_w,  const float* __restrict__ p_b,
    half_t* __restrict__ thetaT, half_t* __restrict__ phiT,
    half_t* __restrict__ Vc) {
    __shared__ float wG[CIN * IC], wT[CIN * IC], wP[CIN * IC];
    __shared__ float bG[IC], bT[IC], bP[IC];
    const int tid = threadIdx.x;
    for (int i = tid; i < CIN * IC; i += 256) {     // transpose: [oc][c]->[c][oc]
        int oc = i >> 6, c = i & 63;
        wG[c * IC + oc] = g_w[i];
        wT[c * IC + oc] = t_w[i];
        wP[c * IC + oc] = p_w[i];
    }
    if (tid < IC) { bG[tid] = g_b[tid]; bT[tid] = t_b[tid]; bP[tid] = p_b[tid]; }
    __syncthreads();

    const int b = blockIdx.y;
    const int n = blockIdx.x * 256 + tid;
    float aG[IC], aT[IC], aP[IC];
#pragma unroll
    for (int o = 0; o < IC; ++o) { aG[o] = 0.f; aT[o] = 0.f; aP[o] = 0.f; }

    const float* xp = x + (size_t)b * CIN * NTOT + n;
    for (int c = 0; c < CIN; ++c) {
        float xv = xp[(size_t)c * NTOT];
#pragma unroll
        for (int j = 0; j < 8; ++j) {
            v4f wg = *(const v4f*)&wG[c * IC + j * 4];
            v4f wt = *(const v4f*)&wT[c * IC + j * 4];
            v4f wp = *(const v4f*)&wP[c * IC + j * 4];
#pragma unroll
            for (int e = 0; e < 4; ++e) {
                aG[j * 4 + e] += wg[e] * xv;
                aT[j * 4 + e] += wt[e] * xv;
                aP[j * 4 + e] += wp[e] * xv;
            }
        }
    }
    v16h t0, t1, p0, p1;
#pragma unroll
    for (int o = 0; o < 16; ++o) {
        t0[o] = (half_t)(aT[o]      + bT[o]);
        t1[o] = (half_t)(aT[o + 16] + bT[o + 16]);
        p0[o] = (half_t)(aP[o]      + bP[o]);
        p1[o] = (half_t)(aP[o + 16] + bP[o + 16]);
    }
    half_t* trow = thetaT + ((size_t)b * NTOT + n) * IC;
    half_t* prow = phiT   + ((size_t)b * NTOT + n) * IC;
    *(v16h*)(trow)      = t0;
    *(v16h*)(trow + 16) = t1;
    *(v16h*)(prow)      = p0;
    *(v16h*)(prow + 16) = p1;
#pragma unroll
    for (int o = 0; o < IC; ++o)
        Vc[((size_t)b * IC + o) * NTOT + n] = (half_t)(aG[o] + bG[o]);
}

// ---------------------------------------------------------------------------
// Streaming-softmax attention phase (flash style), one wave = 16 queries.
//   S^T = K_tile(16x32) @ Q^T(32x16)  -> rows=keys, cols=queries
//   O(cxq) += V^T(16x32keys) @ P^T(32keys x 16q)
// K/V tiles double-buffered in LDS, filled with global_load_async_to_lds_b128
// (ASYNCcnt) so the next tile's copy overlaps the current tile's WMMAs.
// Requires kbIters even (64 and 4 here) for the cross-phase buffer hand-off.
// ---------------------------------------------------------------------------
__device__ __forceinline__ void attn_phase(
    const half_t* __restrict__ Qrow,   // (16 queries x 32) row-major, this wave
    const half_t* __restrict__ Kglob,  // key rows (keys x 32), 2048 halves/64-key blk
    const half_t* __restrict__ Vglob,  // (32 x keys) c-major, row stride vstride
    int vstride, int kbIters, float sc2,  // sc2 = softmax_scale * log2(e)
    float* __restrict__ outBase,       // + c*NTOT + nIdx
    int nIdx, half_t* Kt, half_t* Vt, half_t* myP, int tid) {
    const int lane = tid & 31;
    const int g = lane >> 4;
    const int q = lane & 15;
    const int vc  = tid >> 3, vseg = tid & 7;   // V-tile cooperative mapping

    // Q as WMMA B-fragment: lane holds column q, halves K = 8g+h | 16+8g+h
    v16h Qf;
    ((v8h*)&Qf)[0] = *(const v8h*)(Qrow + q * IC + g * 8);
    ((v8h*)&Qf)[1] = *(const v8h*)(Qrow + q * IC + g * 8 + 16);

    v8f O0 = {0.f,0.f,0.f,0.f,0.f,0.f,0.f,0.f};
    v8f O1 = {0.f,0.f,0.f,0.f,0.f,0.f,0.f,0.f};
    float m2 = -__builtin_inff();
    float l  = 0.f;

    // per-thread async copy of one 16B chunk of K and one of V into buffer buf
    auto issue_tile = [&](int kb, int buf) {
        uint32_t dK = lds_off(Kt + buf * 2048 + tid * 8);
        uint64_t aK = (uint64_t)(uintptr_t)(Kglob + (size_t)kb * 2048 + tid * 8);
        asm volatile("global_load_async_to_lds_b128 %0, %1, off"
                     :: "v"(dK), "v"(aK) : "memory");
        uint32_t dV = lds_off(Vt + buf * 2048 + vc * 64 + vseg * 8);
        uint64_t aV = (uint64_t)(uintptr_t)(Vglob + (size_t)vc * vstride +
                                            (size_t)kb * 64 + vseg * 8);
        asm volatile("global_load_async_to_lds_b128 %0, %1, off"
                     :: "v"(dV), "v"(aV) : "memory");
    };

    for (int kb = 0; kb < kbIters; ++kb) {
        const int cur = kb & 1;
        if (kb == 0) issue_tile(0, 0);
        // my portion of the current tile has landed in LDS
        asm volatile("s_wait_asynccnt 0x0" ::: "memory");
        // everyone's portion landed; also: all waves finished reading buf cur^1
        __syncthreads();
        if (kb + 1 < kbIters) issue_tile(kb + 1, cur ^ 1);

        const half_t* Ktc = Kt + cur * 2048;
        const half_t* Vtc = Vt + cur * 2048;

        // 4 x  S^T subtiles of 16 keys each
        v8f st[4];
#pragma unroll
        for (int j = 0; j < 4; ++j) {
            v16h Af;   // A-fragment: lane row = key, halves K(c) = 8g+h | 16+8g+h
            const half_t* kp = Ktc + (j * 16 + q) * IC + 8 * g;
            ((v8h*)&Af)[0] = *(const v8h*)(kp);
            ((v8h*)&Af)[1] = *(const v8h*)(kp + 16);
            v8f z = {0.f,0.f,0.f,0.f,0.f,0.f,0.f,0.f};
            st[j] = __builtin_amdgcn_wmma_f32_16x16x32_f16(
                false, Af, false, Qf, (short)0, z, false, false);
        }
        // scale to base-2 domain; lane-local max over its 32 (key,q) values
        float mloc = -__builtin_inff();
#pragma unroll
        for (int j = 0; j < 4; ++j)
#pragma unroll
            for (int r = 0; r < 8; ++r) {
                st[j][r] *= sc2;
                mloc = fmaxf(mloc, st[j][r]);
            }
        // combine the two half-lanes holding the same query column
        mloc = fmaxf(mloc, __shfl_xor(mloc, 16, 32));
        float mnew  = fmaxf(m2, mloc);
        float alpha = exp2f(m2 - mnew);
        float psum = 0.f;
#pragma unroll
        for (int j = 0; j < 4; ++j)
#pragma unroll
            for (int r = 0; r < 8; ++r) {
                float p = exp2f(st[j][r] - mnew);
                st[j][r] = p;
                psum += p;
            }
        psum += __shfl_xor(psum, 16, 32);
        l = l * alpha + psum;
        m2 = mnew;
#pragma unroll
        for (int r = 0; r < 8; ++r) { O0[r] *= alpha; O1[r] *= alpha; }

        // stage P^T to per-wave LDS as (q, k) row-major (16 x 64 halves)
#pragma unroll
        for (int j = 0; j < 4; ++j) {
            v8h ph;
#pragma unroll
            for (int r = 0; r < 8; ++r) ph[r] = (half_t)st[j][r];
            *(v8h*)(myP + q * 64 + j * 16 + 8 * g) = ph;   // key = j*16+8g+r
        }
        // PV: two 32-key chunks, two c-tiles
#pragma unroll
        for (int kc = 0; kc < 2; ++kc) {
            // B-fragment of P^T: lane col q, halves k = kc*32 + 8g+h | kc*32+16+8g+h
            v16h Bf;
            ((v8h*)&Bf)[0] = *(const v8h*)(myP + q * 64 + kc * 32 + 8 * g);
            ((v8h*)&Bf)[1] = *(const v8h*)(myP + q * 64 + kc * 32 + 8 * g + 16);
            v16h A0, A1;   // A-fragment of V^T: lane row c, halves k map
            const half_t* v0 = Vtc + q * 64 + kc * 32 + 8 * g;
            const half_t* v1 = Vtc + (16 + q) * 64 + kc * 32 + 8 * g;
            ((v8h*)&A0)[0] = *(const v8h*)(v0);
            ((v8h*)&A0)[1] = *(const v8h*)(v0 + 16);
            ((v8h*)&A1)[0] = *(const v8h*)(v1);
            ((v8h*)&A1)[1] = *(const v8h*)(v1 + 16);
            O0 = __builtin_amdgcn_wmma_f32_16x16x32_f16(
                false, A0, false, Bf, (short)0, O0, false, false);
            O1 = __builtin_amdgcn_wmma_f32_16x16x32_f16(
                false, A1, false, Bf, (short)0, O1, false, false);
        }
    }
    // epilogue: normalize and scatter O(c x q) -> out[c*NTOT + n]
    float inv = 1.f / l;
#pragma unroll
    for (int r = 0; r < 8; ++r) {
        int c0 = 8 * g + r;          // D-fragment rows: M = r + 8g (+16 for tile 1)
        int c1 = 16 + 8 * g + r;
        outBase[(size_t)c0 * NTOT + nIdx] = O0[r] * inv;
        outBase[(size_t)c1 * NTOT + nIdx] = O1[r] * inv;
    }
}

__global__ __launch_bounds__(256) void attn_kernel(
    const half_t* __restrict__ thetaT, const half_t* __restrict__ phiT,
    const half_t* __restrict__ Vc,  const half_t* __restrict__ mbT,
    const half_t* __restrict__ mbC, float* __restrict__ y_ws,
    float* __restrict__ y1_ws) {
    __shared__ half_t Kt[2 * 64 * IC];     // double-buffered K tiles
    __shared__ half_t Vt[2 * IC * 64];     // double-buffered V tiles
    __shared__ half_t Pst[8 * 16 * 64];    // per-wave P^T staging
    const int tid  = threadIdx.x;
    const int wave = tid >> 5;
    const int b    = blockIdx.y;
    const int nb   = blockIdx.x * 128 + wave * 16;
    const int nIdx = nb + (tid & 15);
    half_t* myP = Pst + wave * 1024;
    const float L2E = 1.4426950408889634f;

    // phase 1: non-local self-attention (keys = 4096 spatial positions)
    attn_phase(thetaT + ((size_t)b * NTOT + nb) * IC,
               phiT + (size_t)b * NTOT * IC,
               Vc + (size_t)b * IC * NTOT, NTOT, NTOT / 64, L2E,
               y_ws + (size_t)b * IC * NTOT, nIdx, Kt, Vt, myP, tid);
    // phase 2: memory-bank cross-attention (keys = 256, scale = inter^-0.5)
    attn_phase(phiT + ((size_t)b * NTOT + nb) * IC,
               mbT, mbC, 256, 256 / 64, L2E * 0.17677669529663687f,
               y1_ws + (size_t)b * IC * NTOT, nIdx, Kt, Vt, myP, tid);
}

// ---------------------------------------------------------------------------
// Kernel 3: out = x + W_w @ y + Wz1_w @ y1 + biases   (fp32, coalesced)
// ---------------------------------------------------------------------------
__global__ __launch_bounds__(256) void combine_kernel(
    const float* __restrict__ x,
    const float* __restrict__ W_w,  const float* __restrict__ W_b,
    const float* __restrict__ Z_w,  const float* __restrict__ Z_b,
    const float* __restrict__ y,    const float* __restrict__ y1,
    float* __restrict__ out) {
    __shared__ float wA[IC * CIN], wB[IC * CIN];
    __shared__ float bias[CIN];
    const int tid = threadIdx.x;
    for (int i = tid; i < IC * CIN; i += 256) {   // transpose: [o][ic]->[ic][o]
        int o = i >> 5, ic = i & 31;
        wA[ic * CIN + o] = W_w[i];
        wB[ic * CIN + o] = Z_w[i];
    }
    if (tid < CIN) bias[tid] = W_b[tid] + Z_b[tid];
    __syncthreads();

    const int b = blockIdx.y;
    const int n = blockIdx.x * 256 + tid;
    float acc[CIN];
    const float* xp = x + (size_t)b * CIN * NTOT + n;
#pragma unroll
    for (int o = 0; o < CIN; ++o) acc[o] = xp[(size_t)o * NTOT] + bias[o];

    const float* yp  = y  + (size_t)b * IC * NTOT + n;
    const float* y1p = y1 + (size_t)b * IC * NTOT + n;
    for (int ic = 0; ic < IC; ++ic) {
        float yv  = yp[(size_t)ic * NTOT];
        float y1v = y1p[(size_t)ic * NTOT];
#pragma unroll
        for (int j = 0; j < 16; ++j) {
            v4f wa = *(const v4f*)&wA[ic * CIN + j * 4];
            v4f wb = *(const v4f*)&wB[ic * CIN + j * 4];
#pragma unroll
            for (int e = 0; e < 4; ++e)
                acc[j * 4 + e] += wa[e] * yv + wb[e] * y1v;
        }
    }
    float* op = out + (size_t)b * CIN * NTOT + n;
#pragma unroll
    for (int o = 0; o < CIN; ++o) op[(size_t)o * NTOT] = acc[o];
}

// ---------------------------------------------------------------------------
extern "C" void kernel_launch(void* const* d_in, const int* in_sizes, int n_in,
                              void* d_out, int out_size, void* d_ws, size_t ws_size,
                              hipStream_t stream) {
    const float* x    = (const float*)d_in[0];
    const float* g_w  = (const float*)d_in[1];
    const float* g_b  = (const float*)d_in[2];
    const float* t_w  = (const float*)d_in[3];
    const float* t_b  = (const float*)d_in[4];
    const float* p_w  = (const float*)d_in[5];
    const float* p_b  = (const float*)d_in[6];
    const float* W_w  = (const float*)d_in[7];
    const float* W_b  = (const float*)d_in[8];
    const float* Z_w  = (const float*)d_in[9];
    const float* Z_b  = (const float*)d_in[10];
    const float* mb   = (const float*)d_in[11];

    char* ws = (char*)d_ws;
    const size_t MB = 1u << 20;
    half_t* thetaT = (half_t*)(ws + 0 * MB);        // 2 MB (8*4096*32 f16)
    half_t* phiT   = (half_t*)(ws + 2 * MB);        // 2 MB
    half_t* Vc     = (half_t*)(ws + 4 * MB);        // 2 MB
    half_t* mbT    = (half_t*)(ws + 6 * MB);        // 16 KB (256x32)
    half_t* mbC    = (half_t*)(ws + 6 * MB + 16384);// 16 KB (32x256)
    float*  y_ws   = (float*) (ws + 8 * MB);        // 4 MB (8*32*4096 f32)
    float*  y1_ws  = (float*) (ws + 12 * MB);       // 4 MB

    init_mb_kernel<<<32, 256, 0, stream>>>(mb, mbC, mbT);
    proj_kernel<<<dim3(NTOT / 256, 8), 256, 0, stream>>>(
        x, g_w, g_b, t_w, t_b, p_w, p_b, thetaT, phiT, Vc);
    attn_kernel<<<dim3(NTOT / 128, 8), 256, 0, stream>>>(
        thetaT, phiT, Vc, mbT, mbC, y_ws, y1_ws);
    combine_kernel<<<dim3(NTOT / 256, 8), 256, 0, stream>>>(
        x, W_w, W_b, Z_w, Z_b, y_ws, y1_ws, (float*)d_out);
}
